// SalMM_60232621359320
// MI455X (gfx1250) — compile-verified, matrix operand order checked by reading, not run on hardware
//
#include <hip/hip_runtime.h>
#include <hip/hip_bf16.h>
#include <cstdint>

// ---------------------------------------------------------------------------
// CDNA5 (gfx1250, wave32) SalMM forward skeleton, round 4.
//  * Template-constant shapes, WMMA f32_16x16x32_f16, unrolled K loops,
//    TDM weight DMA into LDS + s_wait_tensorcnt, branch-free epilogues.
//  * This round: unsigned index math (indices are non-negative) so div/mod
//    by powers of two lower to lshr/and instead of signed sign-fixup chains
//    (round-3 asm showed v_ashrrev/v_mul_i24/v_sub sequences per store row).
// ---------------------------------------------------------------------------

typedef __attribute__((ext_vector_type(16))) _Float16 v16h;
typedef __attribute__((ext_vector_type(8)))  float    v8f;
typedef unsigned int v4u  __attribute__((ext_vector_type(4)));
typedef int          v8i_t __attribute__((ext_vector_type(8)));
typedef int          v4i_t __attribute__((ext_vector_type(4)));

// ---- Tensor Data Mover: 1D DMA of `nhalf` f16 elements global -> LDS -------
__device__ __forceinline__ void tdm_load_lds(void* lds_dst, const void* gsrc,
                                             unsigned nhalf) {
  v4u g0; v8i_t g1; v4i_t g2; v4i_t g3;
  unsigned lds_off = (unsigned)(unsigned long long)(uintptr_t)lds_dst;
  unsigned long long ga = (unsigned long long)(uintptr_t)gsrc;
  g0[0] = 1u;                                             // count=1
  g0[1] = lds_off;
  g0[2] = (unsigned)ga;
  g0[3] = ((unsigned)(ga >> 32) & 0x01FFFFFFu) | (2u << 30);   // type=2
  g1[0] = (int)(1u << 16);                                // data_size=2B
  g1[1] = (int)((nhalf & 0xFFFFu) << 16);                 // tensor_dim0 lo
  g1[2] = (int)((nhalf >> 16) & 0xFFFFu);                 // tensor_dim0 hi
  g1[3] = (int)((nhalf & 0xFFFFu) << 16);                 // tile_dim0
  g1[4] = 0;
  g1[5] = (int)nhalf;                                     // dim0 stride
  g1[6] = 0;
  g1[7] = 0;
  g2[0] = g2[1] = g2[2] = g2[3] = 0;
  g3[0] = g3[1] = g3[2] = g3[3] = 0;
#if __clang_major__ >= 23
  v8i_t g4; g4[0]=g4[1]=g4[2]=g4[3]=g4[4]=g4[5]=g4[6]=g4[7]=0;
  __builtin_amdgcn_tensor_load_to_lds(g0, g1, g2, g3, g4, 0);
#else
  __builtin_amdgcn_tensor_load_to_lds(g0, g1, g2, g3, 0);
#endif
  __builtin_amdgcn_s_wait_tensorcnt(0);
}

__device__ __forceinline__ float activate(float v, int act) {
  switch (act) {
    case 1: return fmaxf(v, 0.0f);                                  // relu
    case 2: return 0.5f * v * (1.0f + erff(v * 0.70710678118f));    // gelu
    case 3: return v * (1.0f / (1.0f + __expf(-v)));                // silu
    case 4: return (v > 20.0f) ? v : __logf(1.0f + __expf(v));      // softplus
  }
  return v;
}

// ---------------------------------------------------------------------------
// Pack f32 weight matrix (N x K row-major) -> f16 (Npad x Kpad), zero padded
// ---------------------------------------------------------------------------
__global__ void pack_w_k(const float* __restrict__ w, _Float16* __restrict__ o,
                         unsigned N, unsigned K, unsigned Kpad, unsigned Npad) {
  unsigned i = blockIdx.x * 256 + threadIdx.x;
  if (i >= Npad * Kpad) return;
  unsigned n = i / Kpad, k = i % Kpad;
  o[i] = (_Float16)((n < N && k < K) ? w[n * K + k] : 0.0f);
}

// ---------------------------------------------------------------------------
// WMMA GEMM, all shapes compile-time:  out(M,N) = act(A @ W(N,K)^T + bias)
//   AMODE==0 : A row-major, leading dim LDA
//   AMODE==1 : A is NCHW (m = b*HW + pix, k = channel index, CA channels)
// 256 threads = 8 waves, 128 rows/block (M must be a multiple of 128),
// 16 cols per blockIdx.y tile. Weights DMA'd into LDS by the TDM.
// ---------------------------------------------------------------------------
template <unsigned K, unsigned KPAD, unsigned N, unsigned NPAD, unsigned LDA,
          int ACT, bool HB, int AMODE, unsigned HW, unsigned CA>
__global__ void wmma_gemm_k(const float* __restrict__ A,
                            const _Float16* __restrict__ Wp,
                            const float* __restrict__ bias,
                            float* __restrict__ out) {
  __shared__ __align__(16) _Float16 sA[128 * 32];
  __shared__ __align__(16) _Float16 sB[NPAD * KPAD];
  const unsigned tid  = threadIdx.x;
  const unsigned wave = tid >> 5, lane = tid & 31;
  const unsigned half = lane >> 4, l16 = lane & 15;
  const unsigned m0 = blockIdx.x * 128;
  const unsigned n0 = blockIdx.y * 16;

  if (tid < 32) tdm_load_lds(sB, Wp, NPAD * KPAD);
  __syncthreads();

  v8f acc = {};
#pragma unroll
  for (unsigned k0 = 0; k0 < KPAD; k0 += 32) {
#pragma unroll
    for (unsigned i = tid; i < 128 * 32; i += 256) {
      unsigned r = i >> 5, kk = (i & 31) + k0;
      unsigned m = m0 + r;
      float v = 0.0f;
      if (kk < K) {
        if constexpr (AMODE == 0)
          v = A[(size_t)m * LDA + kk];
        else
          v = A[((size_t)(m / HW) * CA + kk) * HW + (m % HW)];
      }
      sA[r * 32 + (i & 31)] = (_Float16)v;
    }
    __syncthreads();
    v16h af, bf;
#pragma unroll
    for (unsigned v2 = 0; v2 < 8; ++v2) {
      unsigned kk = ((v2 >> 2) * 16) + half * 8 + (v2 & 3) * 2;  // ISA A layout
      af[2 * v2]     = sA[(wave * 16 + l16) * 32 + kk];
      af[2 * v2 + 1] = sA[(wave * 16 + l16) * 32 + kk + 1];
      bf[2 * v2]     = sB[(n0 + l16) * KPAD + k0 + kk];
      bf[2 * v2 + 1] = sB[(n0 + l16) * KPAD + k0 + kk + 1];
    }
    acc = __builtin_amdgcn_wmma_f32_16x16x32_f16(false, af, false, bf,
                                                 (short)0, acc, false, false);
    __syncthreads();
  }
  // ---- epilogue: single column guard, bias hoisted (invariant over r) -----
  const unsigned nn = n0 + l16;
  if (nn < N) {
    float bv = 0.0f;
    if constexpr (HB) bv = bias[nn];
    const unsigned mb = m0 + wave * 16 + half * 8;   // C/D VGPR layout
#pragma unroll
    for (unsigned r = 0; r < 8; ++r)
      out[(size_t)(mb + r) * N + nn] = activate(acc[r] + bv, ACT);
  }
}

// ---------------------------------------------------------------------------
// 3x3 conv (pad=1) as implicit GEMM with WMMA; NCHW f32 in/out.
// All dims compile-time; Bn*HH*WW must be a multiple of 128.
// ---------------------------------------------------------------------------
template <unsigned CIN, unsigned HH, unsigned WW, unsigned COUT,
          unsigned K, unsigned KPAD>
__global__ void conv3x3_wmma_k(const float* __restrict__ x,
                               const _Float16* __restrict__ Wp,
                               const float* __restrict__ bias,
                               float* __restrict__ out) {
  __shared__ __align__(16) _Float16 sA[128 * 32];
  __shared__ __align__(16) _Float16 sB[16 * KPAD];
  const unsigned tid  = threadIdx.x;
  const unsigned wave = tid >> 5, lane = tid & 31;
  const unsigned half = lane >> 4, l16 = lane & 15;
  constexpr unsigned HW = HH * WW;
  const unsigned m0 = blockIdx.x * 128;

  __builtin_prefetch(x + (size_t)m0, 0, 0);   // global_prefetch_b8
  if (tid < 32) tdm_load_lds(sB, Wp, 16 * KPAD);
  __syncthreads();

  v8f acc = {};
#pragma unroll
  for (unsigned k0 = 0; k0 < KPAD; k0 += 32) {
#pragma unroll
    for (unsigned i = tid; i < 128 * 32; i += 256) {
      unsigned r = i >> 5, kk = (i & 31) + k0;
      unsigned m = m0 + r;
      float v = 0.0f;
      if (kk < K) {
        unsigned cin = kk / 9, t = kk % 9, ky = t / 3, kx = t % 3;
        unsigned b = m / HW, rem = m % HW, y = rem / WW, xx = rem % WW;
        int yy = (int)(y + ky) - 1, xw = (int)(xx + kx) - 1;
        if (yy >= 0 && yy < (int)HH && xw >= 0 && xw < (int)WW)
          v = x[((size_t)(b * CIN + cin) * HH + (unsigned)yy) * WW + (unsigned)xw];
      }
      sA[r * 32 + (i & 31)] = (_Float16)v;
    }
    __syncthreads();
    v16h af, bf;
#pragma unroll
    for (unsigned v2 = 0; v2 < 8; ++v2) {
      unsigned kk = ((v2 >> 2) * 16) + half * 8 + (v2 & 3) * 2;
      af[2 * v2]     = sA[(wave * 16 + l16) * 32 + kk];
      af[2 * v2 + 1] = sA[(wave * 16 + l16) * 32 + kk + 1];
      bf[2 * v2]     = sB[l16 * KPAD + k0 + kk];
      bf[2 * v2 + 1] = sB[l16 * KPAD + k0 + kk + 1];
    }
    acc = __builtin_amdgcn_wmma_f32_16x16x32_f16(false, af, false, bf,
                                                 (short)0, acc, false, false);
    __syncthreads();
  }
  // ---- epilogue: guard + bias hoisted out of the row loop -----------------
  if (l16 < COUT) {
    const float bv = bias[l16];
    const unsigned mb = m0 + wave * 16 + half * 8;
#pragma unroll
    for (unsigned r = 0; r < 8; ++r) {
      unsigned m = mb + r;
      unsigned b = m / HW, pix = m % HW;   // lshr/and (unsigned, HW = 2^k)
      out[(size_t)(b * COUT + l16) * HW + pix] = acc[r] + bv;   // NCHW
    }
  }
  // Wave-cluster sync path (spec: NOP when ClusterID==0 / not in a cluster)
  if (__builtin_amdgcn_cluster_id_x() > 0) __builtin_amdgcn_s_cluster_barrier();
}

// ---------------------------------------------------------------------------
// GroupNorm (groups=4), in-place capable. One block per (batch, group).
// ---------------------------------------------------------------------------
template <unsigned C, unsigned HW>
__global__ void groupnorm_k(const float* __restrict__ in,
                            float* __restrict__ out) {
  constexpr unsigned groups = 4, cpg = C / 4, n = cpg * HW;
  unsigned b = blockIdx.x / groups, gr = blockIdx.x % groups;
  size_t base = ((size_t)b * C + (size_t)gr * cpg) * HW;
  float s = 0.0f, s2 = 0.0f;
  for (unsigned i = threadIdx.x; i < n; i += 256) {
    float v = in[base + i]; s += v; s2 += v * v;
  }
  __shared__ float rs[256], rq[256];
  rs[threadIdx.x] = s; rq[threadIdx.x] = s2;
  __syncthreads();
  for (unsigned off = 128; off > 0; off >>= 1) {
    if (threadIdx.x < off) {
      rs[threadIdx.x] += rs[threadIdx.x + off];
      rq[threadIdx.x] += rq[threadIdx.x + off];
    }
    __syncthreads();
  }
  float mean = rs[0] / n;
  float inv  = rsqrtf(rq[0] / n - mean * mean + 1e-5f);
  for (unsigned i = threadIdx.x; i < n; i += 256)
    out[base + i] = (in[base + i] - mean) * inv;   // gamma=1, beta=0
}

// maxpool 2x2 stride 2 then GELU, NCHW
template <unsigned C, unsigned H, unsigned W>
__global__ void pool2_gelu_k(const float* __restrict__ in,
                             float* __restrict__ out, unsigned Bn) {
  constexpr unsigned Ho = H / 2, Wo = W / 2;
  size_t total = (size_t)Bn * C * Ho * Wo;
  size_t i = (size_t)blockIdx.x * 256 + threadIdx.x;
  if (i >= total) return;
  unsigned wo = (unsigned)(i % Wo); size_t t = i / Wo;
  unsigned ho = (unsigned)(t % Ho); t /= Ho;
  unsigned c  = (unsigned)(t % C);  unsigned b = (unsigned)(t / C);
  const float* p = in + ((size_t)(b * C + c) * H + 2 * ho) * W + 2 * wo;
  float m = fmaxf(fmaxf(p[0], p[1]), fmaxf(p[W], p[W + 1]));
  out[i] = 0.5f * m * (1.0f + erff(m * 0.70710678118f));
}

// LayerNorm over last (small, compile-time) dim, identity affine
template <unsigned D>
__global__ void layernorm_k(const float* __restrict__ in,
                            float* __restrict__ out, unsigned M) {
  unsigned m = blockIdx.x * 256 + threadIdx.x;
  if (m >= M) return;
  const float* p = in + (size_t)m * D;
  float s = 0.0f;
#pragma unroll
  for (unsigned d = 0; d < D; ++d) s += p[d];
  float mean = s / D, v = 0.0f;
#pragma unroll
  for (unsigned d = 0; d < D; ++d) { float t = p[d] - mean; v += t * t; }
  float inv = rsqrtf(v / D + 1e-5f);
  float* o = out + (size_t)m * D;
#pragma unroll
  for (unsigned d = 0; d < D; ++d) o[d] = (p[d] - mean) * inv;
}

// Mamba depthwise causal conv (dk=4) + SiLU; rows of xz have stride 2*DI.
template <unsigned DI, unsigned L>
__global__ void dwconv_silu_k(const float* __restrict__ xz,
                              const float* __restrict__ w,
                              const float* __restrict__ bias,
                              float* __restrict__ xc, unsigned Bn) {
  size_t total = (size_t)Bn * L * DI;
  size_t i = (size_t)blockIdx.x * 256 + threadIdx.x;
  if (i >= total) return;
  unsigned d = (unsigned)(i % DI); size_t t = i / DI;
  unsigned l = (unsigned)(t % L);  unsigned b = (unsigned)(t / L);
  float acc = bias[d];
#pragma unroll
  for (int j = 0; j < 4; ++j) {
    int ll = (int)l - 3 + j;
    if (ll >= 0)
      acc += xz[((size_t)b * L + (unsigned)ll) * (2 * DI) + d] * w[d * 4 + j];
  }
  xc[i] = acc * (1.0f / (1.0f + __expf(-acc)));
}

// Mamba selective scan: h[16] in registers, exp on the trans ALU.
template <unsigned DI, unsigned L, unsigned P, unsigned DTR>
__global__ void mamba_scan_k(const float* __restrict__ delta,  // (B*L,DI)
                             const float* __restrict__ proj,   // (B*L,P)
                             const float* __restrict__ xc,     // (B*L,DI)
                             const float* __restrict__ xz,     // (B*L,2DI)
                             const float* __restrict__ A_log,  // (DI,16)
                             const float* __restrict__ Dp,     // (DI)
                             float* __restrict__ y,            // (B*L,DI)
                             unsigned Bn) {
  unsigned idx = blockIdx.x * 256 + threadIdx.x;
  if (idx >= Bn * DI) return;
  unsigned b = idx / DI, d = idx % DI;
  float negA[16], h[16];
#pragma unroll
  for (unsigned s = 0; s < 16; ++s) {
    negA[s] = -__expf(A_log[d * 16 + s]); h[s] = 0.0f;
  }
  for (unsigned l = 0; l < L; ++l) {
    size_t r = (size_t)b * L + l;
    float dt = delta[r * DI + d];
    float xt = xc[r * DI + d];
    const float* pr = proj + r * P;
    float yv = 0.0f;
#pragma unroll
    for (unsigned s = 0; s < 16; ++s) {
      float dA = __expf(dt * negA[s]);
      h[s] = dA * h[s] + dt * pr[DTR + s] * xt;
      yv  += h[s] * pr[DTR + 16 + s];
    }
    float zt = xz[r * 2 * DI + DI + d];
    float sz = zt * (1.0f / (1.0f + __expf(-zt)));
    y[r * DI + d] = (yv + xt * Dp[d]) * sz;
  }
}

// out = a + skip[0]*b   (mamba residual)
__global__ void axpy_k(const float* __restrict__ a, const float* __restrict__ b,
                       const float* __restrict__ skip, float* __restrict__ out,
                       size_t n) {
  size_t i = (size_t)blockIdx.x * 256 + threadIdx.x;
  if (i < n) out[i] = a[i] + skip[0] * b[i];
}

// half-pixel bilinear 2x upsample + sigmoid -> d_out (B,1,2H,2W)
template <unsigned H, unsigned W>
__global__ void up2_sigmoid_k(const float* __restrict__ in,
                              float* __restrict__ out, unsigned Bn,
                              size_t bstride) {
  constexpr unsigned Ho = 2 * H, Wo = 2 * W;
  size_t total = (size_t)Bn * Ho * Wo;
  size_t i = (size_t)blockIdx.x * 256 + threadIdx.x;
  if (i >= total) return;
  unsigned xo = (unsigned)(i % Wo); size_t t = i / Wo;
  unsigned yo = (unsigned)(t % Ho); unsigned b = (unsigned)(t / Ho);
  float fy = (yo + 0.5f) * 0.5f - 0.5f, fx = (xo + 0.5f) * 0.5f - 0.5f;
  int y0 = (int)floorf(fy), x0 = (int)floorf(fx);
  float wy = fy - y0, wx = fx - x0;
  int y1 = min(y0 + 1, (int)H - 1), x1 = min(x0 + 1, (int)W - 1);
  y0 = max(y0, 0); x0 = max(x0, 0);
  const float* p = in + (size_t)b * bstride;
  float v = (1 - wy) * ((1 - wx) * p[y0 * W + x0] + wx * p[y0 * W + x1]) +
            wy       * ((1 - wx) * p[y1 * W + x0] + wx * p[y1 * W + x1]);
  out[i] = 1.0f / (1.0f + __expf(-v));
}

// ---------------------------------------------------------------------------
extern "C" void kernel_launch(void* const* d_in, const int* in_sizes, int n_in,
                              void* d_out, int out_size, void* d_ws, size_t ws_size,
                              hipStream_t stream) {
  constexpr int B = 16, H = 256, W = 256;
  const float* x = (const float*)d_in[0];

  // locate params by element count (flatten order unknown; deterministic)
  auto find = [&](long long sz, int occ) -> const float* {
    for (int i = 0; i < n_in; ++i)
      if ((long long)in_sizes[i] == sz && occ-- == 0) return (const float*)d_in[i];
    return (const float*)d_in[0];
  };
  const float* enc1w = find(8 * 27, 0);   const float* enc1b = find(8, 0);
  const float* enc2w = find(12 * 72, 0);  const float* enc2b = find(12, 0);
  const float* enc3w = find(16 * 108, 0); const float* enc3b = find(16, 0);
  const float* b1w = find(6 * 16, 0);     const float* b1b = find(6, 0);
  const float* m1_in   = find(24 * 6, 0);
  const float* m1_cw   = find(12 * 4, 0); const float* m1_cb = find(12, 1);
  const float* m1_xp   = find(33 * 24, 0);
  const float* m1_dtw  = find(12, 2);     const float* m1_dtb = find(12, 3);
  const float* m1_Alog = find(12 * 16, 0);
  const float* m1_D    = find(12, 4);
  const float* m1_out  = find(6 * 12, 0);
  const float* skip0   = find(1, 0);

  // workspace bump allocator (256B aligned)
  char* wsp = (char*)d_ws; size_t off = 0;
  auto alloc = [&](size_t bytes) -> void* {
    void* p = wsp + off; off = (off + bytes + 255) & ~(size_t)255; return p;
  };
  float* a1 = (float*)alloc((size_t)B * 8  * H     * W     * 4);
  float* p1 = (float*)alloc((size_t)B * 8  * (H/2) * (W/2) * 4);
  float* a2 = (float*)alloc((size_t)B * 12 * (H/2) * (W/2) * 4);
  float* p2 = (float*)alloc((size_t)B * 12 * (H/4) * (W/4) * 4);
  float* a3 = (float*)alloc((size_t)B * 16 * (H/4) * (W/4) * 4);
  float* p3 = (float*)alloc((size_t)B * 16 * (H/8) * (W/8) * 4);
  _Float16* w1p = (_Float16*)alloc(16 * 32  * 2);
  _Float16* w2p = (_Float16*)alloc(16 * 96  * 2);
  _Float16* w3p = (_Float16*)alloc(16 * 128 * 2);
  _Float16* wb1 = (_Float16*)alloc(16 * 32  * 2);
  _Float16* wip = (_Float16*)alloc(32 * 32  * 2);
  _Float16* wxp = (_Float16*)alloc(48 * 32  * 2);
  _Float16* wdt = (_Float16*)alloc(16 * 32  * 2);
  _Float16* wop = (_Float16*)alloc(16 * 32  * 2);
  const int M4 = B * 1024;                      // enc4 sequence: 16 x (32*32)
  float* s1    = (float*)alloc((size_t)M4 * 6  * 4);
  float* s1n   = (float*)alloc((size_t)M4 * 6  * 4);
  float* xz    = (float*)alloc((size_t)M4 * 24 * 4);
  float* xc    = (float*)alloc((size_t)M4 * 12 * 4);
  float* proj  = (float*)alloc((size_t)M4 * 33 * 4);
  float* delta = (float*)alloc((size_t)M4 * 12 * 4);
  float* ys    = (float*)alloc((size_t)M4 * 12 * 4);
  float* mo    = (float*)alloc((size_t)M4 * 6  * 4);
  (void)ws_size;

  auto cdiv = [](long long a, long long b) { return (int)((a + b - 1) / b); };

  // ---- pack weights to f16 (Npad x Kpad) ----------------------------------
  pack_w_k<<<cdiv(16 * 32, 256),  256, 0, stream>>>(enc1w, w1p, 8, 27, 32, 16);
  pack_w_k<<<cdiv(16 * 96, 256),  256, 0, stream>>>(enc2w, w2p, 12, 72, 96, 16);
  pack_w_k<<<cdiv(16 * 128, 256), 256, 0, stream>>>(enc3w, w3p, 16, 108, 128, 16);
  pack_w_k<<<cdiv(16 * 32, 256),  256, 0, stream>>>(b1w,   wb1, 6, 16, 32, 16);
  pack_w_k<<<cdiv(32 * 32, 256),  256, 0, stream>>>(m1_in, wip, 24, 6, 32, 32);
  pack_w_k<<<cdiv(48 * 32, 256),  256, 0, stream>>>(m1_xp, wxp, 33, 24, 32, 48);
  pack_w_k<<<cdiv(16 * 32, 256),  256, 0, stream>>>(m1_dtw, wdt, 12, 1, 32, 16);
  pack_w_k<<<cdiv(16 * 32, 256),  256, 0, stream>>>(m1_out, wop, 6, 12, 32, 16);

  // ---- encoder stem: conv3x3 -> GN -> pool+GELU (x3) ----------------------
  int M1 = B * H * W;                 // 1048576, multiple of 128
  conv3x3_wmma_k<3, 256, 256, 8, 27, 32>
      <<<M1 / 128, 256, 0, stream>>>(x, w1p, enc1b, a1);
  groupnorm_k<8, 256 * 256><<<B * 4, 256, 0, stream>>>(a1, a1);
  pool2_gelu_k<8, 256, 256>
      <<<cdiv((long long)B * 8 * 128 * 128, 256), 256, 0, stream>>>(a1, p1, B);

  int M2 = B * (H / 2) * (W / 2);     // 262144
  conv3x3_wmma_k<8, 128, 128, 12, 72, 96>
      <<<M2 / 128, 256, 0, stream>>>(p1, w2p, enc2b, a2);
  groupnorm_k<12, 128 * 128><<<B * 4, 256, 0, stream>>>(a2, a2);
  pool2_gelu_k<12, 128, 128>
      <<<cdiv((long long)B * 12 * 64 * 64, 256), 256, 0, stream>>>(a2, p2, B);

  int M3 = B * (H / 4) * (W / 4);     // 65536
  conv3x3_wmma_k<12, 64, 64, 16, 108, 128>
      <<<M3 / 128, 256, 0, stream>>>(p2, w3p, enc3b, a3);
  groupnorm_k<16, 64 * 64><<<B * 4, 256, 0, stream>>>(a3, a3);
  pool2_gelu_k<16, 64, 64>
      <<<cdiv((long long)B * 16 * 32 * 32, 256), 256, 0, stream>>>(a3, p3, B);

  // ---- SCP enc4 branch 1 + Mamba m1 (M4 = 16384, multiple of 128) ---------
  // b1 = relu(1x1 conv): NCHW gather GEMM (K=16 -> N=6)
  {
    dim3 g(M4 / 128, 1);
    wmma_gemm_k<16, 32, 6, 16, 16, 1, true, 1, 1024, 16>
        <<<g, 256, 0, stream>>>(p3, wb1, b1b, s1);
  }
  layernorm_k<6><<<cdiv(M4, 256), 256, 0, stream>>>(s1, s1n, M4);
  // in_proj: (M,6) -> (M,24)
  {
    dim3 g(M4 / 128, 2);
    wmma_gemm_k<6, 32, 24, 32, 6, 0, false, 0, 1, 1>
        <<<g, 256, 0, stream>>>(s1n, wip, nullptr, xz);
  }
  dwconv_silu_k<12, 1024>
      <<<cdiv((long long)M4 * 12, 256), 256, 0, stream>>>(xz, m1_cw, m1_cb, xc, B);
  // x_proj: (M,12) -> (M,33)  [dt | B | C]
  {
    dim3 g(M4 / 128, 3);
    wmma_gemm_k<12, 32, 33, 48, 12, 0, false, 0, 1, 1>
        <<<g, 256, 0, stream>>>(xc, wxp, nullptr, proj);
  }
  // delta = softplus(dt @ dt_w^T + dt_b): K=1, LDA=33 selects the dt column
  {
    dim3 g(M4 / 128, 1);
    wmma_gemm_k<1, 32, 12, 16, 33, 4, true, 0, 1, 1>
        <<<g, 256, 0, stream>>>(proj, wdt, m1_dtb, delta);
  }
  mamba_scan_k<12, 1024, 33, 1>
      <<<1, 256, 0, stream>>>(delta, proj, xc, xz, m1_Alog, m1_D, ys, B);
  // out_proj: (M,12) -> (M,6)
  {
    dim3 g(M4 / 128, 1);
    wmma_gemm_k<12, 32, 6, 16, 12, 0, false, 0, 1, 1>
        <<<g, 256, 0, stream>>>(ys, wop, nullptr, mo);
  }
  axpy_k<<<cdiv((long long)M4 * 6, 256), 256, 0, stream>>>(
      mo, s1n, skip0, mo, (size_t)M4 * 6);

  // ---- head: 2x bilinear upsample + sigmoid into d_out --------------------
  up2_sigmoid_k<256, 256>
      <<<cdiv((long long)B * 512 * 512, 256), 256, 0, stream>>>(
          a1, (float*)d_out, B, (size_t)8 * H * W);
  (void)out_size;
}
